// ResidualIrregularConv1D_28346784154089
// MI455X (gfx1250) — matrix operand order repaired and checked by reading, compiled, not run on hardware
//
#include <hip/hip_runtime.h>
#include <hip/hip_bf16.h>
#include <stdint.h>

// ---------------------------------------------------------------------------
// Types for CDNA5 WMMA
// ---------------------------------------------------------------------------
typedef __attribute__((ext_vector_type(16))) __bf16 v16bf;
typedef __attribute__((ext_vector_type(8)))  __bf16 v8bf;
typedef __attribute__((ext_vector_type(8)))  float  v8f;
typedef __attribute__((ext_vector_type(4)))  int    v4i;

#define EPS 1e-3f
#define E_DIM 1024
#define M_DIM 8192   // B*S = 512*16

// float -> bf16 (round-to-nearest-even), stored as u16
__device__ __forceinline__ unsigned short f2bf(float f) {
    union { float f; uint32_t u; } v; v.f = f;
    uint32_t u = v.u;
    u += 0x7fffu + ((u >> 16) & 1u);
    return (unsigned short)(u >> 16);
}

// ---------------------------------------------------------------------------
// Global -> LDS 16B copy: async path if toolchain exposes it, else plain copy
// ---------------------------------------------------------------------------
#if defined(__gfx1250__) && __has_builtin(__builtin_amdgcn_global_load_async_to_lds_b128)
#define USE_ASYNC_LDS 1
#else
#define USE_ASYNC_LDS 0
#endif

typedef __attribute__((address_space(1))) v4i* gas_v4i_ptr;
typedef __attribute__((address_space(3))) v4i* las_v4i_ptr;

__device__ __forceinline__ void cp_g2l_16(const void* g, void* l) {
#if USE_ASYNC_LDS
    __builtin_amdgcn_global_load_async_to_lds_b128(
        (gas_v4i_ptr)(g),
        (las_v4i_ptr)(l), 0, 0);
#else
    *(uint4*)l = *(const uint4*)g;
#endif
}

__device__ __forceinline__ void stage_wait() {
#if USE_ASYNC_LDS
#if __has_builtin(__builtin_amdgcn_s_wait_asynccnt)
    __builtin_amdgcn_s_wait_asynccnt(0);
#else
    asm volatile("s_wait_asynccnt 0" ::: "memory");
#endif
#endif
    __syncthreads();
}

// ---------------------------------------------------------------------------
// Prep: fold BN params into per-column scale/shift, build Wp_eff from masks
// ---------------------------------------------------------------------------
__global__ void rc1d_prep_kernel(
    const float* __restrict__ Wp, const float* __restrict__ bp,
    const float* __restrict__ b1, const float* __restrict__ b2,
    const float* __restrict__ g1, const float* __restrict__ be1,
    const float* __restrict__ m1, const float* __restrict__ v1,
    const float* __restrict__ g2, const float* __restrict__ be2,
    const float* __restrict__ m2, const float* __restrict__ v2,
    const float* __restrict__ g3, const float* __restrict__ be3,
    const float* __restrict__ m3, const float* __restrict__ v3,
    const int*   __restrict__ masks,
    float* __restrict__ WpEff,
    float* __restrict__ sc1, float* __restrict__ sh1,
    float* __restrict__ sc2, float* __restrict__ sh2,
    float* __restrict__ sc3, float* __restrict__ sh3)
{
    const int t = threadIdx.x;           // 256 threads
    // Wp_eff[r][s] = sum over m with masks[m]%16 == r of Wp[m][s]
    const int r = t >> 4, s = t & 15;
    float acc = 0.f;
    for (int m = 0; m < 64; ++m) {
        int rr = masks[m] & 15;
        if (rr == r) acc += Wp[m * 16 + s];
    }
    WpEff[r * 16 + s] = acc;
    if (t < 16) {
        float sc = g1[t] * rsqrtf(v1[t] + EPS);
        sc1[t] = sc;
        sh1[t] = (bp[t] - m1[t]) * sc + be1[t];
    }
    for (int c = t; c < E_DIM; c += 256) {
        float s2 = g2[c] * rsqrtf(v2[c] + EPS);
        sc2[c] = s2;
        sh2[c] = (b1[c] - m2[c]) * s2 + be2[c];
        float s3 = g3[c] * rsqrtf(v3[c] + EPS);
        sc3[c] = s3;
        sh3[c] = (b2[c] - m3[c]) * s3 + be3[c];
    }
}

// ---------------------------------------------------------------------------
// Transpose + convert W (K x N f32, row-major) -> Wt (N x K bf16, row-major)
// ---------------------------------------------------------------------------
__global__ __launch_bounds__(256) void rc1d_transpose_kernel(
    const float* __restrict__ W, unsigned short* __restrict__ Wt, int N)
{
    __shared__ float tile[32][33];
    const int bx = blockIdx.x * 32;      // n block
    const int by = blockIdx.y * 32;      // k block
    const int tx = threadIdx.x & 31;
    const int ty = threadIdx.x >> 5;     // 0..7
    #pragma unroll
    for (int r = ty; r < 32; r += 8)
        tile[r][tx] = W[(size_t)(by + r) * N + bx + tx];
    __syncthreads();
    #pragma unroll
    for (int r = ty; r < 32; r += 8)
        Wt[(size_t)(bx + r) * N + by + tx] = f2bf(tile[tx][r]);
}

// ---------------------------------------------------------------------------
// Stage 1: Z[b,s,e] = relu(bn1(sum_r X[b,r,e] * WpEff[r,s])) -> bf16 (8192x1024)
// ---------------------------------------------------------------------------
__global__ __launch_bounds__(256) void rc1d_stage1_kernel(
    const float* __restrict__ X, const float* __restrict__ WpEff,
    const float* __restrict__ sc1, const float* __restrict__ sh1,
    unsigned short* __restrict__ Z)
{
    __shared__ float w[256];
    __shared__ float s1[16], h1[16];
    const int t = threadIdx.x;
    w[t] = WpEff[t];
    if (t < 16) { s1[t] = sc1[t]; h1[t] = sh1[t]; }
    __syncthreads();

    const int b = blockIdx.y;
    const int e = blockIdx.x * 256 + t;
    float x[16];
    #pragma unroll
    for (int r = 0; r < 16; ++r)
        x[r] = X[((size_t)b * 16 + r) * E_DIM + e];
    #pragma unroll
    for (int s = 0; s < 16; ++s) {
        float p = 0.f;
        #pragma unroll
        for (int r = 0; r < 16; ++r) p = fmaf(x[r], w[r * 16 + s], p);
        float z = fmaxf(fmaf(p, s1[s], h1[s]), 0.f);
        Z[((size_t)b * 16 + s) * E_DIM + e] = f2bf(z);
    }
}

// ---------------------------------------------------------------------------
// Tiled bf16 GEMM with fused BN epilogue.
//   C(M,N) = A(M,K) @ B(K,N);  B supplied pre-transposed: Bt(N,K)
//   epilogue: val = acc*scale[n] + shift[n]
//     final==0: out_bf16 = bf16(relu(val))
//     final==1: out_f32  = relu(Xres + val)
// Block tile 128x128, 8 waves of 32 (2Mx4N), wave tile 64x32 = 4x2 WMMA tiles,
// K-step 32, double-buffered LDS (80B row pitch -> conflict-free b128 reads).
// ---------------------------------------------------------------------------
#define BM 128
#define BN 128
#define BK 32
#define LROW 80            // bytes per LDS row (32 bf16 + 16B pad)
#define LTILE (128 * LROW) // 10240 B per tile

__global__ __launch_bounds__(256) void rc1d_gemm_bn_kernel(
    const unsigned short* __restrict__ A,   // M x K bf16
    const unsigned short* __restrict__ Bt,  // N x K bf16
    const float* __restrict__ scale,        // N
    const float* __restrict__ shift,        // N
    const float* __restrict__ Xres,         // M x N f32 (final only)
    unsigned short* __restrict__ outbf,     // M x N bf16 (final==0)
    float* __restrict__ outf,               // M x N f32  (final==1)
    int M, int N, int K, int final_stage)
{
    __shared__ __align__(16) unsigned char smem[2][2][LTILE];

    const int tid  = threadIdx.x;
    const int lane = tid & 31;
    const int w    = tid >> 5;       // 0..7
    const int wm   = w >> 2;         // 0..1  -> 64 rows each
    const int wn   = w & 3;          // 0..3  -> 32 cols each
    const int m0   = blockIdx.y * BM;
    const int n0   = blockIdx.x * BN;
    const int rsel = lane & 15;
    const int hsel = lane >> 4;      // half-select per ISA 16-bit layouts

    v8f acc[4][2];
    #pragma unroll
    for (int i = 0; i < 4; ++i)
        #pragma unroll
        for (int j = 0; j < 2; ++j)
            acc[i][j] = (v8f){0.f,0.f,0.f,0.f,0.f,0.f,0.f,0.f};

    auto stage = [&](int buf, int k0) {
        #pragma unroll
        for (int pass = 0; pass < 2; ++pass) {
            int c   = pass * 256 + tid;       // 0..511
            int row = c >> 2;                 // 0..127
            int off = (c & 3) * 16;           // 0/16/32/48 bytes
            cp_g2l_16((const char*)A  + ((size_t)(m0 + row) * K + k0) * 2 + off,
                      &smem[buf][0][row * LROW + off]);
            cp_g2l_16((const char*)Bt + ((size_t)(n0 + row) * K + k0) * 2 + off,
                      &smem[buf][1][row * LROW + off]);
        }
    };

    const int NT = K / BK;   // 32
    stage(0, 0);
    stage_wait();

    for (int kt = 0; kt < NT; ++kt) {
        const int cur = kt & 1;
        if (kt + 1 < NT) stage(cur ^ 1, (kt + 1) * BK);

        const unsigned char* sA = smem[cur][0];
        const unsigned char* sB = smem[cur][1];

        // A fragments: row M=rsel; lanes 0-15 hold K0..7,K16..23; lanes 16-31 K8..15,K24..31
        v16bf a[4], b[2];
        const int o0 = hsel * 16;
        #pragma unroll
        for (int i = 0; i < 4; ++i) {
            const unsigned char* p = sA + (wm * 64 + i * 16 + rsel) * LROW;
            ((v8bf*)&a[i])[0] = *(const v8bf*)(p + o0);
            ((v8bf*)&a[i])[1] = *(const v8bf*)(p + o0 + 32);
        }
        // B fragments: col N=rsel; lanes 0-15 hold K0..7,K16..23; lanes 16-31 K8..15,K24..31
        #pragma unroll
        for (int j = 0; j < 2; ++j) {
            const unsigned char* p = sB + (wn * 32 + j * 16 + rsel) * LROW + hsel * 16;
            ((v8bf*)&b[j])[0] = *(const v8bf*)(p);
            ((v8bf*)&b[j])[1] = *(const v8bf*)(p + 32);
        }
        #pragma unroll
        for (int i = 0; i < 4; ++i)
            #pragma unroll
            for (int j = 0; j < 2; ++j)
                acc[i][j] = __builtin_amdgcn_wmma_f32_16x16x32_bf16(
                    false, a[i], false, b[j], (short)0, acc[i][j], false, false);

        stage_wait();
    }

    // Epilogue. C layout: VGPR v -> M = v + 8*hsel, N = rsel (per lane)
    #pragma unroll
    for (int i = 0; i < 4; ++i) {
        const int rbase = m0 + wm * 64 + i * 16 + hsel * 8;
        #pragma unroll
        for (int j = 0; j < 2; ++j) {
            const int gcol = n0 + wn * 32 + j * 16 + rsel;
            const float sc = scale[gcol];
            const float sh = shift[gcol];
            #pragma unroll
            for (int v = 0; v < 8; ++v) {
                const size_t idx = (size_t)(rbase + v) * N + gcol;
                float val = fmaf(acc[i][j][v], sc, sh);
                if (final_stage) {
                    outf[idx] = fmaxf(Xres[idx] + val, 0.f);
                } else {
                    outbf[idx] = f2bf(fmaxf(val, 0.f));
                }
            }
        }
    }
}

// ---------------------------------------------------------------------------
// Launch
// ---------------------------------------------------------------------------
extern "C" void kernel_launch(void* const* d_in, const int* in_sizes, int n_in,
                              void* d_out, int out_size, void* d_ws, size_t ws_size,
                              hipStream_t stream) {
    const float* X   = (const float*)d_in[0];
    const float* Wp  = (const float*)d_in[1];
    const float* bp  = (const float*)d_in[2];
    const float* W1  = (const float*)d_in[3];
    const float* b1  = (const float*)d_in[4];
    const float* W2  = (const float*)d_in[5];
    const float* b2  = (const float*)d_in[6];
    const float* g1  = (const float*)d_in[7];
    const float* be1 = (const float*)d_in[8];
    const float* m1  = (const float*)d_in[9];
    const float* v1  = (const float*)d_in[10];
    const float* g2  = (const float*)d_in[11];
    const float* be2 = (const float*)d_in[12];
    const float* m2  = (const float*)d_in[13];
    const float* v2  = (const float*)d_in[14];
    const float* g3  = (const float*)d_in[15];
    const float* be3 = (const float*)d_in[16];
    const float* m3  = (const float*)d_in[17];
    const float* v3  = (const float*)d_in[18];
    const int* masks = (const int*)d_in[19];
    float* out = (float*)d_out;

    char* ws = (char*)d_ws;
    unsigned short* Zbuf = (unsigned short*)(ws);                        // 16 MiB
    unsigned short* Hbuf = (unsigned short*)(ws + (16u << 20));          // 16 MiB
    unsigned short* Wt1  = (unsigned short*)(ws + (32u << 20));          // 2 MiB
    unsigned short* Wt2  = (unsigned short*)(ws + (34u << 20));          // 2 MiB
    float* WpEff = (float*)(ws + (36u << 20));                           // 1 KiB
    float* sc1   = WpEff + 256;
    float* sh1   = sc1 + 16;
    float* sc2   = (float*)(ws + (36u << 20) + 4096);
    float* sh2   = sc2 + 1024;
    float* sc3   = sh2 + 1024;
    float* sh3   = sc3 + 1024;

    rc1d_prep_kernel<<<1, 256, 0, stream>>>(
        Wp, bp, b1, b2, g1, be1, m1, v1, g2, be2, m2, v2, g3, be3, m3, v3,
        masks, WpEff, sc1, sh1, sc2, sh2, sc3, sh3);

    rc1d_transpose_kernel<<<dim3(32, 32), 256, 0, stream>>>(W1, Wt1, E_DIM);
    rc1d_transpose_kernel<<<dim3(32, 32), 256, 0, stream>>>(W2, Wt2, E_DIM);

    rc1d_stage1_kernel<<<dim3(4, 512), 256, 0, stream>>>(X, WpEff, sc1, sh1, Zbuf);

    dim3 ggrid(E_DIM / BN, M_DIM / BM);   // (8, 64)
    rc1d_gemm_bn_kernel<<<ggrid, 256, 0, stream>>>(
        Zbuf, Wt1, sc2, sh2, nullptr, Hbuf, nullptr, M_DIM, E_DIM, E_DIM, 0);
    rc1d_gemm_bn_kernel<<<ggrid, 256, 0, stream>>>(
        Hbuf, Wt2, sc3, sh3, X, nullptr, out, M_DIM, E_DIM, E_DIM, 1);
}
